// MambaBlock2D_27453430956042
// MI455X (gfx1250) — compile-verified, matrix operand order checked by reading, not run on hardware
//
#include <hip/hip_runtime.h>
#include <math.h>

typedef __attribute__((ext_vector_type(16))) __bf16 v16bf;
typedef __attribute__((ext_vector_type(8)))  __bf16 v8bf;
typedef __attribute__((ext_vector_type(8)))  float  v8f;

#define BB   2
#define CC   96
#define HH   96
#define WWD  96
#define LL   (HH*WWD)        // 9216
#define BCL  (BB*CC*LL)      // 1,769,472
#define K4   4
#define NN   8
#define RR   6
#define DDIM 22              // R + 2N
#define XPM  32              // xproj M rows padded to 32
#define NTL  576             // LL/16
#define MTL  6               // CC/16
#define CHK  96              // scan chunks
#define SSC  (LL/CHK)        // 96
#define PB   256             // partial-reduction blocks per batch
#define C8   12              // CC/8

// fast transcendental helpers: v_exp_f32 + v_rcp_f32, no IEEE-div ladder
__device__ __forceinline__ float sigm(float v) {
  return __builtin_amdgcn_rcpf(1.f + __expf(-v));
}
__device__ __forceinline__ float siluf(float v)    { return v * sigm(v); }
__device__ __forceinline__ float softplusf(float v){ return v > 20.f ? v : __logf(1.f + __expf(v)); }

__device__ __forceinline__ v16bf cat16(v8bf lo, v8bf hi) {
  return __builtin_shufflevector(lo, hi, 0,1,2,3,4,5,6,7,8,9,10,11,12,13,14,15);
}

// ---------------- deterministic two-stage batch reduction (mean/var) -------
__global__ __launch_bounds__(256) void stats_partial(const float* __restrict__ src,
                                                     float* __restrict__ part, int M) {
  int b  = blockIdx.x / PB;
  int pb = blockIdx.x % PB;
  const float* p = src + (size_t)b * M;
  float s = 0.f, s2 = 0.f;
  for (int i = pb * 256 + threadIdx.x; i < M; i += PB * 256) {
    float v = p[i]; s += v; s2 += v * v;
  }
  __shared__ float sh[256], sh2[256];
  sh[threadIdx.x] = s; sh2[threadIdx.x] = s2;
  __syncthreads();
  for (int o = 128; o > 0; o >>= 1) {
    if (threadIdx.x < o) { sh[threadIdx.x] += sh[threadIdx.x + o]; sh2[threadIdx.x] += sh2[threadIdx.x + o]; }
    __syncthreads();
  }
  if (threadIdx.x == 0) {
    part[(b * PB + pb) * 2 + 0] = sh[0];
    part[(b * PB + pb) * 2 + 1] = sh2[0];
  }
}

__global__ __launch_bounds__(256) void stats_final(const float* __restrict__ part,
                                                   float* __restrict__ murs, int M) {
  __shared__ float sh[256], sh2[256];
  for (int b = 0; b < BB; ++b) {
    sh[threadIdx.x]  = part[(b * PB + threadIdx.x) * 2 + 0];
    sh2[threadIdx.x] = part[(b * PB + threadIdx.x) * 2 + 1];
    __syncthreads();
    for (int o = 128; o > 0; o >>= 1) {
      if (threadIdx.x < o) { sh[threadIdx.x] += sh[threadIdx.x + o]; sh2[threadIdx.x] += sh2[threadIdx.x + o]; }
      __syncthreads();
    }
    if (threadIdx.x == 0) {
      float mu  = sh[0] / (float)M;
      float var = sh2[0] / (float)M - mu * mu;
      murs[2 * b + 0] = mu;
      murs[2 * b + 1] = rsqrtf(var + 1e-5f);
    }
    __syncthreads();
  }
}

// ---------------- one-shot weight conversion to bf16 -----------------------
// p1/p2: 96x96 row-major.  xproj: per-k 22x96 padded to 32x96 (zeros).
__global__ __launch_bounds__(256) void k_prep_w(const float* __restrict__ p1w,
    const float* __restrict__ p2w, const float* __restrict__ xpw,
    __bf16* __restrict__ w1bf, __bf16* __restrict__ w2bf, __bf16* __restrict__ wxbf) {
  int i = blockIdx.x * 256 + threadIdx.x;
  if (i < CC * CC) { w1bf[i] = (__bf16)p1w[i]; return; }
  i -= CC * CC;
  if (i < CC * CC) { w2bf[i] = (__bf16)p2w[i]; return; }
  i -= CC * CC;
  if (i < K4 * XPM * CC) {
    int c = i % CC; int d = (i / CC) % XPM; int k = i / (XPM * CC);
    wxbf[i] = (d < DDIM) ? (__bf16)xpw[((size_t)k * DDIM + d) * CC + c] : (__bf16)0.0f;
  }
}

// ---------------- GN1 apply, emit channel-major bf16 for WMMA B-operand ----
// hbf_t layout: [(b*LL + l)*CC + c]  (C contiguous, 192B row pitch, 16B aligned)
__global__ __launch_bounds__(256) void k_gn_apply_bf16(const float* __restrict__ x,
    const float* __restrict__ murs, const float* __restrict__ g,
    const float* __restrict__ bt, __bf16* __restrict__ hbf_t) {
  int i = blockIdx.x * 256 + threadIdx.x;
  if (i >= BCL) return;
  int l = i % LL;
  int c = (i / LL) % CC;
  int b = i / (CC * LL);
  float v = (x[i] - murs[2 * b]) * murs[2 * b + 1] * g[c] + bt[c];
  hbf_t[((size_t)b * LL + l) * CC + c] = (__bf16)v;
}

// ---------------- fused p1/p2 1x1 conv: bf16 WMMA, f32 accumulate ----------
// One wave32 computes one 16x16 output tile; K=96 in three k-steps of 32.
// All four fragment chunks are contiguous 16-byte global_load_b128's.
__global__ __launch_bounds__(256) void k_gemm_p1p2(const __bf16* __restrict__ hbf_t,
    const __bf16* __restrict__ w1bf, const __bf16* __restrict__ w2bf,
    const float* __restrict__ p1b, const float* __restrict__ p2b,
    float* __restrict__ abuf, float* __restrict__ bpre) {
  int wg   = blockIdx.x * 8 + (threadIdx.x >> 5);
  int lane = threadIdx.x & 31;
  int half = lane >> 4, lm = lane & 15;
  int h8   = half * 8;
  int nt    = wg % NTL;
  int mt    = (wg / NTL) % MTL;
  int b     = (wg / (NTL * MTL)) % BB;
  int which = wg / (NTL * MTL * BB);
  const __bf16* wmat = which ? w2bf : w1bf;
  const float*  bias = which ? p2b : p1b;
  int col = nt * 16 + lm;          // l index
  int row = mt * 16 + lm;          // output channel (A row)
  size_t abase = (size_t)row * CC + h8;
  size_t bbase = ((size_t)b * LL + col) * CC + h8;
  v8f acc = {0.f, 0.f, 0.f, 0.f, 0.f, 0.f, 0.f, 0.f};
#pragma unroll
  for (int ks = 0; ks < 3; ++ks) {
    int kb = ks * 32;
    if (ks < 2) __builtin_prefetch(hbf_t + bbase + kb + 32, 0, 3);
    v8bf alo = *(const v8bf*)(wmat  + abase + kb);
    v8bf ahi = *(const v8bf*)(wmat  + abase + kb + 16);
    v8bf blo = *(const v8bf*)(hbf_t + bbase + kb);
    v8bf bhi = *(const v8bf*)(hbf_t + bbase + kb + 16);
    acc = __builtin_amdgcn_wmma_f32_16x16x32_bf16(false, cat16(alo, ahi),
                                                  false, cat16(blo, bhi),
                                                  (short)0, acc, false, false);
  }
#pragma unroll
  for (int r = 0; r < 8; ++r) {
    int o = mt * 16 + r + h8;        // C/D layout: VGPR r -> row M
    float v = acc[r] + bias[o];
    size_t oi = (size_t)(b * CC + o) * LL + col;
    if (which) bpre[oi] = v; else abuf[oi] = siluf(v);
  }
}

// ---------------- depthwise 3x3 conv + bias + SiLU -------------------------
__global__ __launch_bounds__(256) void k_dwconv(const float* __restrict__ bpre,
    const float* __restrict__ dww, const float* __restrict__ dwb,
    float* __restrict__ bconv) {
  int i = blockIdx.x * 256 + threadIdx.x;
  if (i >= BCL) return;
  int w = i % WWD; int h = (i / WWD) % HH;
  int c = (i / LL) % CC; int b = i / (CC * LL);
  const float* wp = dww + c * 9;
  const float* src = bpre + (size_t)(b * CC + c) * LL;
  float acc = 0.f;
#pragma unroll
  for (int dy = -1; dy <= 1; ++dy) {
    int hy = h + dy; if (hy < 0 || hy >= HH) continue;
#pragma unroll
    for (int dx = -1; dx <= 1; ++dx) {
      int wx = w + dx; if (wx < 0 || wx >= WWD) continue;
      acc += wp[(dy + 1) * 3 + (dx + 1)] * src[hy * WWD + wx];
    }
  }
  bconv[i] = siluf(acc + dwb[c]);
}

// ---------------- build 4-direction xs, channel-major bf16 -----------------
// xs_t layout: [((b*K4+k)*LL + l)*CC + c]
__global__ __launch_bounds__(256) void k_build_xs(const float* __restrict__ bconv,
                                                  __bf16* __restrict__ xs_t) {
  int i = blockIdx.x * 256 + threadIdx.x;
  if (i >= BB * K4 * CC * LL) return;
  int l = i % LL; int c = (i / LL) % CC;
  int k = (i / (LL * CC)) % K4; int b = i / (LL * CC * K4);
  int lsrc;
  if (k == 0)      lsrc = l;
  else if (k == 1) lsrc = (l % HH) * WWD + (l / HH);          // (w,h) enumeration
  else if (k == 2) lsrc = LL - 1 - l;
  else { int lr = LL - 1 - l; lsrc = (lr % HH) * WWD + (lr / HH); }
  xs_t[((size_t)(b * K4 + k) * LL + l) * CC + c] =
      (__bf16)bconv[(size_t)(b * CC + c) * LL + lsrc];
}

// ---------------- x-projection GEMM: (22->32)x96 @ 96xL per (b,k), WMMA ----
__global__ __launch_bounds__(256) void k_gemm_xproj(const __bf16* __restrict__ xs_t,
    const __bf16* __restrict__ wxbf, float* __restrict__ xdbl) {
  int wg   = blockIdx.x * 8 + (threadIdx.x >> 5);
  int lane = threadIdx.x & 31;
  int half = lane >> 4, lm = lane & 15;
  int h8   = half * 8;
  int nt = wg % NTL;
  int mt = (wg / NTL) & 1;            // 2 M-tiles (rows 0-15, 16-31; >=22 zero)
  int k  = (wg / (NTL * 2)) % K4;
  int b  = wg / (NTL * 2 * K4);
  int col = nt * 16 + lm;
  int row = mt * 16 + lm;             // d index (padded weights, no guard)
  size_t abase = ((size_t)k * XPM + row) * CC + h8;
  size_t bbase = ((size_t)(b * K4 + k) * LL + col) * CC + h8;
  v8f acc = {0.f, 0.f, 0.f, 0.f, 0.f, 0.f, 0.f, 0.f};
#pragma unroll
  for (int ks = 0; ks < 3; ++ks) {
    int kb = ks * 32;
    if (ks < 2) __builtin_prefetch(xs_t + bbase + kb + 32, 0, 3);
    v8bf alo = *(const v8bf*)(wxbf + abase + kb);
    v8bf ahi = *(const v8bf*)(wxbf + abase + kb + 16);
    v8bf blo = *(const v8bf*)(xs_t + bbase + kb);
    v8bf bhi = *(const v8bf*)(xs_t + bbase + kb + 16);
    acc = __builtin_amdgcn_wmma_f32_16x16x32_bf16(false, cat16(alo, ahi),
                                                  false, cat16(blo, bhi),
                                                  (short)0, acc, false, false);
  }
#pragma unroll
  for (int r = 0; r < 8; ++r) {
    int d = mt * 16 + r + h8;
    if (d < DDIM)
      xdbl[(size_t)((b * K4 + k) * DDIM + d) * LL + col] = acc[r];
  }
}

// ---------------- selective scan: 3-pass chunked decomposition -------------
// pass 1: per (b,k,c,chunk) compute per-n (P = prod dA, F = local state)
__global__ __launch_bounds__(256) void k_scan_pass1(const float* __restrict__ xdbl,
    const __bf16* __restrict__ xs_t, const float* __restrict__ dtw,
    const float* __restrict__ dtb, const float* __restrict__ alog,
    float* __restrict__ PF) {
  int t = blockIdx.x * 256 + threadIdx.x;
  if (t >= BB * K4 * CC * CHK) return;
  int ch = t % CHK; int c = (t / CHK) % CC;
  int k  = (t / (CHK * CC)) % K4; int b = t / (CHK * CC * K4);
  int bk = b * K4 + k;
  float wr[RR];
#pragma unroll
  for (int r = 0; r < RR; ++r) wr[r] = dtw[((size_t)k * CC + c) * RR + r];
  float bconst = dtb[k * CC + c];
  float An[NN];
#pragma unroll
  for (int n = 0; n < NN; ++n) An[n] = -__expf(alog[((size_t)k * CC + c) * NN + n]);
  float P[NN], F[NN];
#pragma unroll
  for (int n = 0; n < NN; ++n) { P[n] = 1.f; F[n] = 0.f; }
  const float*  xb  = xdbl + (size_t)bk * DDIM * LL;
  const __bf16* xsp = xs_t + (size_t)bk * LL * CC + c;
  for (int l = ch * SSC; l < ch * SSC + SSC; ++l) {
    float dtr = bconst;
#pragma unroll
    for (int r = 0; r < RR; ++r) dtr += wr[r] * xb[(size_t)r * LL + l];
    float dt = softplusf(dtr);
    float xv = (float)xsp[(size_t)l * CC];
#pragma unroll
    for (int n = 0; n < NN; ++n) {
      float Bn = xb[(size_t)(RR + n) * LL + l];
      float dA = __expf(dt * An[n]);
      P[n] *= dA;
      F[n]  = dA * F[n] + dt * Bn * xv;
    }
  }
#pragma unroll
  for (int n = 0; n < NN; ++n) {
    PF[(size_t)t * 2 * NN + n]      = P[n];
    PF[(size_t)t * 2 * NN + NN + n] = F[n];
  }
}

// pass 2: sequential cross-chunk combine per (b,k,c); writes per-chunk h_init
__global__ __launch_bounds__(256) void k_scan_pass2(const float* __restrict__ PF,
                                                    float* __restrict__ hinit) {
  int t = blockIdx.x * 256 + threadIdx.x;
  if (t >= BB * K4 * CC) return;
  float h[NN];
#pragma unroll
  for (int n = 0; n < NN; ++n) h[n] = 0.f;
  for (int ch = 0; ch < CHK; ++ch) {
    size_t tt = (size_t)t * CHK + ch;   // same index as pass-1 thread id
#pragma unroll
    for (int n = 0; n < NN; ++n) hinit[tt * NN + n] = h[n];
#pragma unroll
    for (int n = 0; n < NN; ++n)
      h[n] = PF[tt * 2 * NN + n] * h[n] + PF[tt * 2 * NN + NN + n];
  }
}

// pass 3: replay each chunk from h_init, produce y = sum_n C_n h_n + D*x
__global__ __launch_bounds__(256) void k_scan_pass3(const float* __restrict__ xdbl,
    const __bf16* __restrict__ xs_t, const float* __restrict__ dtw,
    const float* __restrict__ dtb, const float* __restrict__ alog,
    const float* __restrict__ Ds, const float* __restrict__ hinit,
    float* __restrict__ ys) {
  int t = blockIdx.x * 256 + threadIdx.x;
  if (t >= BB * K4 * CC * CHK) return;
  int ch = t % CHK; int c = (t / CHK) % CC;
  int k  = (t / (CHK * CC)) % K4; int b = t / (CHK * CC * K4);
  int bk = b * K4 + k;
  float wr[RR];
#pragma unroll
  for (int r = 0; r < RR; ++r) wr[r] = dtw[((size_t)k * CC + c) * RR + r];
  float bconst = dtb[k * CC + c];
  float Dv = Ds[k * CC + c];
  float An[NN], h[NN];
#pragma unroll
  for (int n = 0; n < NN; ++n) {
    An[n] = -__expf(alog[((size_t)k * CC + c) * NN + n]);
    h[n]  = hinit[(size_t)t * NN + n];
  }
  const float*  xb  = xdbl + (size_t)bk * DDIM * LL;
  const __bf16* xsp = xs_t + (size_t)bk * LL * CC + c;
  float* yp = ys + (size_t)(bk * CC + c) * LL;
  for (int l = ch * SSC; l < ch * SSC + SSC; ++l) {
    float dtr = bconst;
#pragma unroll
    for (int r = 0; r < RR; ++r) dtr += wr[r] * xb[(size_t)r * LL + l];
    float dt = softplusf(dtr);
    float xv = (float)xsp[(size_t)l * CC];
    float y  = Dv * xv;
#pragma unroll
    for (int n = 0; n < NN; ++n) {
      float Bn = xb[(size_t)(RR + n) * LL + l];
      float Cn = xb[(size_t)(RR + NN + n) * LL + l];
      float dA = __expf(dt * An[n]);
      h[n] = dA * h[n] + dt * Bn * xv;
      y   += Cn * h[n];
    }
    yp[l] = y;
  }
}

// ---------------- recombine the 4 directions -------------------------------
__global__ __launch_bounds__(256) void k_combine(const float* __restrict__ ys,
                                                 float* __restrict__ ssout) {
  int i = blockIdx.x * 256 + threadIdx.x;
  if (i >= BCL) return;
  int l = i % LL; int c = (i / LL) % CC; int b = i / (CC * LL);
  int h = l / WWD, w = l % WWD;
  size_t s0 = (size_t)((b * K4 + 0) * CC + c) * LL;
  size_t s1 = (size_t)((b * K4 + 1) * CC + c) * LL;
  size_t s2 = (size_t)((b * K4 + 2) * CC + c) * LL;
  size_t s3 = (size_t)((b * K4 + 3) * CC + c) * LL;
  int lt = w * HH + h;
  ssout[i] = ys[s0 + l] + ys[s1 + lt] + ys[s2 + (LL - 1 - l)] + ys[s3 + (LL - 1 - lt)];
}

// ---------------- GN2 apply * a, plus per-(b,c) spatial mean ---------------
__global__ __launch_bounds__(256) void k_gn2_mul_pool(const float* __restrict__ ss,
    const float* __restrict__ murs, const float* __restrict__ g,
    const float* __restrict__ bt, const float* __restrict__ a,
    float* __restrict__ ybuf, float* __restrict__ s) {
  int bc = blockIdx.x; int b = bc / CC; int c = bc % CC;
  size_t base = (size_t)bc * LL;
  float mu = murs[2 * b], rs = murs[2 * b + 1];
  float gg = g[c], bb = bt[c];
  float acc = 0.f;
  for (int l = threadIdx.x; l < LL; l += 256) {
    float bn = (ss[base + l] - mu) * rs * gg + bb;
    float yv = a[base + l] * bn;
    ybuf[base + l] = yv;
    acc += yv;
  }
  __shared__ float sh[256];
  sh[threadIdx.x] = acc;
  __syncthreads();
  for (int o = 128; o > 0; o >>= 1) {
    if (threadIdx.x < o) sh[threadIdx.x] += sh[threadIdx.x + o];
    __syncthreads();
  }
  if (threadIdx.x == 0) s[bc] = sh[0] / (float)LL;
}

// ---------------- channel-attention MLP (tiny) -----------------------------
__global__ __launch_bounds__(256) void k_chattn(const float* __restrict__ s,
    const float* __restrict__ w1, const float* __restrict__ b1,
    const float* __restrict__ w2, const float* __restrict__ b2,
    float* __restrict__ attn) {
  __shared__ float z[BB * C8];
  int t = threadIdx.x;
  if (t < BB * C8) {
    int b = t / C8, j = t % C8;
    float acc = b1[j];
    for (int c = 0; c < CC; ++c) acc += s[b * CC + c] * w1[j * CC + c];
    z[t] = fmaxf(acc, 0.f);
  }
  __syncthreads();
  if (t < BB * CC) {
    int b = t / CC, o = t % CC;
    float acc = b2[o];
#pragma unroll
    for (int j = 0; j < C8; ++j) acc += z[b * C8 + j] * w2[o * C8 + j];
    attn[t] = sigm(acc);
  }
}

// ---------------- residual output ------------------------------------------
__global__ __launch_bounds__(256) void k_final(const float* __restrict__ x,
    const float* __restrict__ ybuf, const float* __restrict__ attn,
    float* __restrict__ out) {
  int i = blockIdx.x * 256 + threadIdx.x;
  if (i >= BCL) return;
  int c = (i / LL) % CC; int b = i / (CC * LL);
  out[i] = x[i] + ybuf[i] * attn[b * CC + c];
}

extern "C" void kernel_launch(void* const* d_in, const int* in_sizes, int n_in,
                              void* d_out, int out_size, void* d_ws, size_t ws_size,
                              hipStream_t stream) {
  (void)in_sizes; (void)n_in; (void)out_size; (void)ws_size;
  const float* x    = (const float*)d_in[0];
  const float* ln_g = (const float*)d_in[1];
  const float* ln_b = (const float*)d_in[2];
  const float* p1w  = (const float*)d_in[3];
  const float* p1b  = (const float*)d_in[4];
  const float* p2w  = (const float*)d_in[5];
  const float* p2b  = (const float*)d_in[6];
  const float* dww  = (const float*)d_in[7];
  const float* dwb  = (const float*)d_in[8];
  const float* xpw  = (const float*)d_in[9];
  const float* dtw  = (const float*)d_in[10];
  const float* dtb  = (const float*)d_in[11];
  const float* alog = (const float*)d_in[12];
  const float* Dsp  = (const float*)d_in[13];
  const float* n2g  = (const float*)d_in[14];
  const float* n2b  = (const float*)d_in[15];
  const float* caw1 = (const float*)d_in[16];
  const float* cab1 = (const float*)d_in[17];
  const float* caw2 = (const float*)d_in[18];
  const float* cab2 = (const float*)d_in[19];
  float* out = (float*)d_out;

  char* ws = (char*)d_ws;
  size_t off = 0;
  auto alloc = [&](size_t bytes) -> char* {
    char* p = ws + off; off += (bytes + 255) & ~(size_t)255; return p;
  };
  __bf16* hbf_t = (__bf16*)alloc((size_t)BCL * 2);
  float*  abuf  = (float*) alloc((size_t)BCL * 4);
  float*  bpre  = (float*) alloc((size_t)BCL * 4);
  float*  bconv = (float*) alloc((size_t)BCL * 4);
  __bf16* xs_t  = (__bf16*)alloc((size_t)BB * K4 * CC * LL * 2);
  float*  xdbl  = (float*) alloc((size_t)BB * K4 * DDIM * LL * 4);
  float*  ys    = (float*) alloc((size_t)BB * K4 * CC * LL * 4);
  float*  ssout = (float*) alloc((size_t)BCL * 4);
  float*  ybuf  = (float*) alloc((size_t)BCL * 4);
  float*  PF    = (float*) alloc((size_t)BB * K4 * CC * CHK * 2 * NN * 4);
  float*  hinit = (float*) alloc((size_t)BB * K4 * CC * CHK * NN * 4);
  float*  part  = (float*) alloc((size_t)BB * PB * 2 * 4);
  float*  murs  = (float*) alloc(256);
  float*  murs2 = (float*) alloc(256);
  float*  sbuf  = (float*) alloc((size_t)BB * CC * 4);
  float*  attn  = (float*) alloc((size_t)BB * CC * 4);
  __bf16* w1bf  = (__bf16*)alloc((size_t)CC * CC * 2);
  __bf16* w2bf  = (__bf16*)alloc((size_t)CC * CC * 2);
  __bf16* wxbf  = (__bf16*)alloc((size_t)K4 * XPM * CC * 2);

  const int M = CC * LL;
  const int NPREP = 2 * CC * CC + K4 * XPM * CC;
  k_prep_w     <<<(NPREP + 255) / 256, 256, 0, stream>>>(p1w, p2w, xpw, w1bf, w2bf, wxbf);
  stats_partial<<<BB * PB, 256, 0, stream>>>(x, part, M);
  stats_final  <<<1, 256, 0, stream>>>(part, murs, M);
  k_gn_apply_bf16<<<(BCL + 255) / 256, 256, 0, stream>>>(x, murs, ln_g, ln_b, hbf_t);
  k_gemm_p1p2 <<<(BB * 2 * MTL * NTL) / 8, 256, 0, stream>>>(hbf_t, w1bf, w2bf, p1b, p2b, abuf, bpre);
  k_dwconv    <<<(BCL + 255) / 256, 256, 0, stream>>>(bpre, dww, dwb, bconv);
  k_build_xs  <<<(BB * K4 * CC * LL + 255) / 256, 256, 0, stream>>>(bconv, xs_t);
  k_gemm_xproj<<<(BB * K4 * 2 * NTL) / 8, 256, 0, stream>>>(xs_t, wxbf, xdbl);
  k_scan_pass1<<<(BB * K4 * CC * CHK + 255) / 256, 256, 0, stream>>>(xdbl, xs_t, dtw, dtb, alog, PF);
  k_scan_pass2<<<(BB * K4 * CC + 255) / 256, 256, 0, stream>>>(PF, hinit);
  k_scan_pass3<<<(BB * K4 * CC * CHK + 255) / 256, 256, 0, stream>>>(xdbl, xs_t, dtw, dtb, alog, Dsp, hinit, ys);
  k_combine   <<<(BCL + 255) / 256, 256, 0, stream>>>(ys, ssout);
  stats_partial<<<BB * PB, 256, 0, stream>>>(ssout, part, M);
  stats_final  <<<1, 256, 0, stream>>>(part, murs2, M);
  k_gn2_mul_pool<<<BB * CC, 256, 0, stream>>>(ssout, murs2, n2g, n2b, abuf, ybuf, sbuf);
  k_chattn    <<<1, 256, 0, stream>>>(sbuf, caw1, cab1, caw2, cab2, attn);
  k_final     <<<(BCL + 255) / 256, 256, 0, stream>>>(x, ybuf, attn, out);
}
